// LSTM_Backbone_3006477107489
// MI455X (gfx1250) — matrix operand order repaired and checked by reading, not compiled, not run on hardware
//
#include <hip/hip_runtime.h>
#include <hip/hip_bf16.h>

// ---------------------------------------------------------------------------
// Bidirectional 2-layer LSTM for MI455X (gfx1250).
//  - bf16 WMMA (v_wmma_f32_16x16x32_bf16) for all GEMMs, fp32 accumulate
//  - gate-interleaved packed weights in exact 16-bit A-fragment lane layout
//  - recurrence: 1 block per (batch-group of 16, direction); h in LDS (bf16),
//    c in VGPRs; two barriers per timestep
//  - 512 threads (16 wave32s) so the per-wave VGPR budget avoids spills;
//    K-loop unrolled by 4 to bound fragment live ranges
//  - branch-free sigmoid/tanh via hardware v_exp_f32 (pointwise tail is on the
//    sequential critical path)
// ---------------------------------------------------------------------------

typedef __attribute__((ext_vector_type(16))) __bf16 v16bf;
typedef __attribute__((ext_vector_type(8)))  float  v8f;

#define B_SZ   64
#define T_SZ   512
#define EMB_SZ 256
#define HID_SZ 512
#define GATES  2048   // 4*HID

__device__ __forceinline__ float sigf(float x) {
  return 1.0f / (1.0f + __expf(-x));
}
__device__ __forceinline__ float tanh_fast(float x) {
  // tanh(x) = 1 - 2/(1 + e^{2x}); __expf saturates to +inf / 0 => +/-1 limits
  return 1.0f - 2.0f / (1.0f + __expf(2.0f * x));
}

// ---------------------------------------------------------------------------
// Pack W = [Wih | Whh] (rows reordered as 4*hid+gate) into WMMA A-fragment
// layout: element ((mt*KT + kt)*32 + lane)*16 + j, where in this code's
// 16-bit A 16x32 layout (contiguous K, matching the B fragments):
//   lane <16 : row M=lane,   j -> K offsets {0..15}
//   lane>=16 : row M=lane-16,j -> K offsets {16..31}
// Also fold bih+bhh into a reordered bias vector.
// ---------------------------------------------------------------------------
__global__ void pack_weights_kernel(const float* __restrict__ Wih,
                                    const float* __restrict__ Whh,
                                    const float* __restrict__ bih,
                                    const float* __restrict__ bhh,
                                    __bf16* __restrict__ packed,
                                    float*  __restrict__ biasR,
                                    int in_dim) {
  const int  K  = in_dim + HID_SZ;
  const int  KT = K >> 5;
  const long total = (long)GATES * K;
  for (long e = (long)blockIdx.x * blockDim.x + threadIdx.x; e < total;
       e += (long)gridDim.x * blockDim.x) {
    const int  j    = (int)(e & 15);
    const int  lane = (int)((e >> 4) & 31);
    const long rest = e >> 9;
    const int  kt   = (int)(rest % KT);
    const int  mt   = (int)(rest / KT);
    const int  mp   = mt * 16 + (lane & 15);          // interleaved row 4*hid+gate
    const int  hid  = mp >> 2, gate = mp & 3;
    const int  morig = gate * HID_SZ + hid;           // reference row (i,f,g,o blocks)
    const int  koff = (lane < 16) ? j
                                  : (j + 16);
    const int  k = kt * 32 + koff;
    const float v = (k < in_dim) ? Wih[(long)morig * in_dim + k]
                                 : Whh[(long)morig * HID_SZ + (k - in_dim)];
    packed[e] = (__bf16)v;
    if (e < GATES) {
      const int h2 = (int)e >> 2, g2 = (int)e & 3;
      biasR[e] = bih[g2 * HID_SZ + h2] + bhh[g2 * HID_SZ + h2];
    }
  }
}

// ---------------------------------------------------------------------------
// Fused recurrent scan for one layer, both directions (blockIdx.y = dir).
// blockIdx.x = batch group of 16.  512 threads = 16 waves; wave w owns the
// 8 gate-row tiles mt = 8w..8w+7 (=> hidden units 32w..32w+31 after the
// 4*hid+gate interleave), keeping c for them in VGPRs across all timesteps.
// ---------------------------------------------------------------------------
template<int IN_DIM>
__global__ __launch_bounds__(512)
void lstm_scan_kernel(const __bf16* __restrict__ packedW,  // [dir][GATES*K]
                      const float*  __restrict__ biasR,    // [dir][GATES]
                      const int*    __restrict__ tokens,
                      const float*  __restrict__ emb,
                      const float*  __restrict__ xin,
                      const int*    __restrict__ lengths,
                      float*        __restrict__ out) {
  constexpr int K  = IN_DIM + HID_SZ;
  constexpr int KT = K >> 5;
  constexpr int RS = K + 16;  // row stride (bf16): 32B multiple, 8-bank skew

  __shared__ __attribute__((aligned(32))) __bf16 r[16][RS]; // [batch][x ; h]
  __shared__ float ldsBias[GATES];

  const int  tid    = threadIdx.x;
  const int  lane   = tid & 31;
  const int  wave   = tid >> 5;
  const int  laneLo = lane & 15;
  const bool hi     = lane >= 16;
  const int  dir    = blockIdx.y;
  const int  n0     = blockIdx.x * 16;

  const __bf16* Wd = packedW + (size_t)dir * GATES * K;
  const float*  Bd = biasR   + (size_t)dir * GATES;

  for (int e = tid; e < 16 * HID_SZ; e += 512)
    r[e >> 9][IN_DIM + (e & 511)] = (__bf16)0.0f;          // h = 0
  for (int e = tid; e < GATES; e += 512) ldsBias[e] = Bd[e];

  float creg[16];
  #pragma unroll
  for (int i2 = 0; i2 < 16; ++i2) creg[i2] = 0.0f;         // c = 0
  const int lenReg = lengths[n0 + laneLo];
  __syncthreads();

  float hn[16];

  for (int s = 0; s < T_SZ; ++s) {
    const int t = dir ? (T_SZ - 1 - s) : s;

    // Phase 1: stage x_t (embedding gather for layer 0) into LDS as bf16
    if (IN_DIM == 256) {
      for (int e = tid; e < 16 * 256; e += 512) {
        const int n = e >> 8, k = e & 255;
        const int tok = tokens[(n0 + n) * T_SZ + t];
        r[n][k] = (__bf16)emb[(size_t)tok * 256 + k];
      }
    } else {
      for (int e = tid; e < 16 * 1024; e += 512) {
        const int n = e >> 10, k = e & 1023;
        r[n][k] = (__bf16)xin[((size_t)(n0 + n) * T_SZ + t) * 1024 + k];
      }
    }
    __syncthreads();

    const bool msk = t < lenReg;
    // B-fragment base: lane n<16 -> col n, K{0..15}; lane>=16 -> col n-16, K{16..31}
    const __bf16* bbase = &r[laneLo][hi ? 16 : 0];

    // Phase 2: gates = W*[x;h] + b (WMMA), then pointwise LSTM cell
    #pragma unroll
    for (int tt = 0; tt < 8; ++tt) {
      const int mt = wave * 8 + tt;
      v8f acc;
      const int brow = mt * 16 + (hi ? 8 : 0);
      #pragma unroll
      for (int v = 0; v < 8; ++v) acc[v] = ldsBias[brow + v];

      const __bf16* aptr = Wd + ((size_t)mt * KT * 32 + lane) * 16;
      __builtin_prefetch((const void*)(aptr + (size_t)KT * 512), 0, 1); // next tile's A
      #pragma unroll 4
      for (int kt = 0; kt < KT; ++kt) {
        v16bf af = *(const v16bf*)(aptr + (size_t)kt * 512);
        v16bf bf = *(const v16bf*)(bbase + kt * 32);
        acc = __builtin_amdgcn_wmma_f32_16x16x32_bf16(
            false, af, false, bf, (short)0, acc, false, false);
      }
      // D-tile rows: lanes<16 hold rows 0..7 (hid mt*4+{0,1}), lanes>=16 rows 8..15
      const float cA = creg[2 * tt], cB = creg[2 * tt + 1];
      const float iA = sigf(acc[0]), fA = sigf(acc[1]);
      const float gA = tanh_fast(acc[2]), oA = sigf(acc[3]);
      const float iB = sigf(acc[4]), fB = sigf(acc[5]);
      const float gB = tanh_fast(acc[6]), oB = sigf(acc[7]);
      const float cnA = fA * cA + iA * gA;
      const float cnB = fB * cB + iB * gB;
      const float hA = oA * tanh_fast(cnA);
      const float hB = oB * tanh_fast(cnB);
      if (msk) { creg[2 * tt] = cnA; creg[2 * tt + 1] = cnB; }
      hn[2 * tt] = hA; hn[2 * tt + 1] = hB;
    }
    __syncthreads();  // all WMMA reads of r[h] done before anyone rewrites h

    // Phase 3: commit h (bf16, feeds next step's B-fragments) and outputs
    const size_t obase =
        ((size_t)(n0 + laneLo) * T_SZ + t) * 1024 + (size_t)dir * HID_SZ;
    #pragma unroll
    for (int tt = 0; tt < 8; ++tt) {
      const int mt   = wave * 8 + tt;
      const int hidA = mt * 4 + (hi ? 2 : 0);
      const float hA = hn[2 * tt], hB = hn[2 * tt + 1];
      if (msk) {
        r[laneLo][IN_DIM + hidA]     = (__bf16)hA;
        r[laneLo][IN_DIM + hidA + 1] = (__bf16)hB;
      }
      out[obase + hidA]     = msk ? hA : 0.0f;
      out[obase + hidA + 1] = msk ? hB : 0.0f;
    }
    // no extra barrier: phase-1 of next step writes only the x region, and its
    // __syncthreads orders these h-writes before the next WMMA reads
  }
}

// ---------------------------------------------------------------------------
// Host launch
// d_in: [0]=inputs(B,T) i32, [1]=lengths(B) i32, [2]=embedding(VOCAB,EMB) f32,
//       then per (layer, dir=fwd..bwd): Wih, Whh, bih, bhh  (f32)
// ws:   x_mid (B*T*1024 f32) | packed bf16 weights (4 cfgs) | reordered biases
// ---------------------------------------------------------------------------
extern "C" void kernel_launch(void* const* d_in, const int* in_sizes, int n_in,
                              void* d_out, int out_size, void* d_ws, size_t ws_size,
                              hipStream_t stream) {
  (void)in_sizes; (void)n_in; (void)out_size; (void)ws_size;
  const int*   tokens  = (const int*)d_in[0];
  const int*   lengths = (const int*)d_in[1];
  const float* emb     = (const float*)d_in[2];

  char*  ws    = (char*)d_ws;
  float* x_mid = (float*)ws;
  size_t off   = (size_t)B_SZ * T_SZ * 1024 * sizeof(float);
  __bf16* packed = (__bf16*)(ws + off);
  const size_t pk_l0 = (size_t)GATES * 768;   // elements per direction, layer 0
  const size_t pk_l1 = (size_t)GATES * 1536;  // elements per direction, layer 1
  float* biasR = (float*)(ws + off + (2 * pk_l0 + 2 * pk_l1) * sizeof(__bf16));

  for (int l = 0; l < 2; ++l) {
    for (int d = 0; d < 2; ++d) {
      const int c = l * 2 + d;
      const float* Wih = (const float*)d_in[3 + c * 4 + 0];
      const float* Whh = (const float*)d_in[3 + c * 4 + 1];
      const float* bih = (const float*)d_in[3 + c * 4 + 2];
      const float* bhh = (const float*)d_in[3 + c * 4 + 3];
      const size_t poff = (c == 0) ? 0
                        : (c == 1) ? pk_l0
                        : (c == 2) ? 2 * pk_l0
                                   : 2 * pk_l0 + pk_l1;
      const int in_dim = (l == 0) ? 256 : 1024;
      hipLaunchKernelGGL(pack_weights_kernel, dim3(4096), dim3(256), 0, stream,
                         Wih, Whh, bih, bhh, packed + poff, biasR + c * GATES,
                         in_dim);
    }
  }

  const dim3 grid(4, 2), block(512);
  hipLaunchKernelGGL((lstm_scan_kernel<256>), grid, block, 0, stream,
                     packed, biasR, tokens, emb, (const float*)nullptr,
                     lengths, x_mid);
  hipLaunchKernelGGL((lstm_scan_kernel<1024>), grid, block, 0, stream,
                     packed + 2 * pk_l0, biasR + 2 * GATES, tokens, emb, x_mid,
                     lengths, (float*)d_out);
}
